// MultiHeadSelfAttention_46153718562844
// MI455X (gfx1250) — matrix operand order repaired and checked by reading, not compiled, not run on hardware
//
#include <hip/hip_runtime.h>
#include <hip/hip_bf16.h>

// ---------------------------------------------------------------------------
// MHSA for B=2, S=4096, D=768, H=12 (DH=64) on gfx1250 (CDNA5, wave32).
// bf16 WMMA (v_wmma_f32_16x16x32_bf16) everywhere, f32 accumulation.
// Flash-attention style: never materialize the S x S score matrix.
// K-chunk + bf16 GEMM A-tile staging use GLOBAL_LOAD_ASYNC_TO_LDS (ASYNCcnt).
// f32 tiles staged as float4 loads + packed ds_store_b64.
// ---------------------------------------------------------------------------

typedef __attribute__((ext_vector_type(16))) __bf16 v16bf;
typedef __attribute__((ext_vector_type(8)))  float  v8f;

constexpr int BATCH  = 2;
constexpr int SEQ    = 4096;
constexpr int DMODEL = 768;
constexpr int NHEAD  = 12;
constexpr int HDIM   = 64;
constexpr int MROWS  = BATCH * SEQ;           // 8192
constexpr int NQKV   = 3 * DMODEL;            // 2304
constexpr float ATT_SCALE = 0.125f;           // 1/sqrt(64)
constexpr float LOG2E     = 1.4426950408889634f;

union FragU { v16bf v; uint4 q[2]; };
union U4H   { uint4 q; unsigned short h[8]; };
union H4U   { unsigned short h[4]; unsigned long long u; };

__device__ __forceinline__ unsigned short f2bf(float f) {
    unsigned u = __builtin_bit_cast(unsigned, f);
    u += 0x7FFFu + ((u >> 16) & 1u);          // round-to-nearest-even
    return (unsigned short)(u >> 16);
}

// Pack 4 f32 -> 4 bf16 in a 64-bit lane value.
__device__ __forceinline__ unsigned long long pack4bf(float4 f) {
    H4U p;
    p.h[0] = f2bf(f.x); p.h[1] = f2bf(f.y);
    p.h[2] = f2bf(f.z); p.h[3] = f2bf(f.w);
    return p.u;
}

// Async copy of 16 bytes global -> LDS (per lane). Low 32 bits of a flat
// shared-memory address are the LDS byte offset (ISA 10.2 aperture rule).
__device__ __forceinline__ void async_copy_b128(void* lds_dst, const void* gsrc) {
    unsigned               laddr = (unsigned)(size_t)lds_dst;
    unsigned long long     gaddr = (unsigned long long)(size_t)gsrc;
    asm volatile("global_load_async_to_lds_b128 %0, %1, off"
                 :: "v"(laddr), "v"(gaddr) : "memory");
}
__device__ __forceinline__ void wait_asynccnt0() {
    asm volatile("s_wait_asynccnt 0x0" ::: "memory");
}

// A fragment (16x32 bf16, MxK). Lane L: M = L&15; K runs = {kb..kb+7, kb+16..kb+23},
// kb = (L>=16 ? 8 : 0). LDS tile is row-major [m][k], stride multiple of 8.
__device__ __forceinline__ v16bf frag_a_lds(const unsigned short* p, int row0,
                                            int koff, int stride, int lane) {
    const unsigned short* r =
        p + (size_t)(row0 + (lane & 15)) * stride + koff + ((lane >> 4) ? 8 : 0);
    FragU f;
    f.q[0] = *(const uint4*)(r);
    f.q[1] = *(const uint4*)(r + 16);
    return f.v;
}

// B fragment (32x16 bf16, KxN). Lane L: N = L&15; K runs = {kb..kb+7, kb+16..kb+23},
// kb = (L>=16 ? 8 : 0). LDS tile stored [n][k] row-major.
__device__ __forceinline__ v16bf frag_b_lds(const unsigned short* p, int n0,
                                            int koff, int stride, int lane) {
    const unsigned short* r =
        p + (size_t)(n0 + (lane & 15)) * stride + koff + ((lane >> 4) ? 8 : 0);
    FragU f;
    f.q[0] = *(const uint4*)(r);
    f.q[1] = *(const uint4*)(r + 16);
    return f.v;
}

#define WMMA_BF16(a, b, c) \
    __builtin_amdgcn_wmma_f32_16x16x32_bf16(false, (a), false, (b), (short)0, (c), false, false)

// ---------------------------------------------------------------------------
// Kernel 1: qkv = x @ W_qkv^T + b_qkv, scattered to Q/K/V [B,H,S,64] bf16.
// Q is pre-scaled by ATT_SCALE*LOG2E so attention can use exp2 directly.
// Block tile 128x128, 8 waves, each wave 32(M) x 64(N).
// ---------------------------------------------------------------------------
__global__ __launch_bounds__(256) void qkv_gemm_kernel(
    const float* __restrict__ x, const float* __restrict__ W,
    const float* __restrict__ bias,
    unsigned short* __restrict__ Qb, unsigned short* __restrict__ Kb,
    unsigned short* __restrict__ Vb) {
    __shared__ alignas(16) unsigned short As[128 * 40];
    __shared__ alignas(16) unsigned short Bs[128 * 40];

    const int tid = threadIdx.x, lane = tid & 31, wave = tid >> 5;
    const int m0 = blockIdx.x * 128, n0 = blockIdx.y * 128;
    const int wm = (wave & 3) * 32, wn = (wave >> 2) * 64;
    const int sr = tid >> 3, sc = (tid & 7) * 4;   // staging row/col (per 32 rows)

    const v8f vzero = {0.f, 0.f, 0.f, 0.f, 0.f, 0.f, 0.f, 0.f};
    v8f acc[2][4];
#pragma unroll
    for (int i = 0; i < 2; ++i)
#pragma unroll
        for (int j = 0; j < 4; ++j) acc[i][j] = vzero;

    for (int k0 = 0; k0 < DMODEL; k0 += 32) {
#pragma unroll
        for (int i = 0; i < 4; ++i) {   // 128x32 f32 -> bf16, float4 + b64 stores
            int r = i * 32 + sr;
            float4 xa = *(const float4*)(&x[(size_t)(m0 + r) * DMODEL + k0 + sc]);
            float4 wa = *(const float4*)(&W[(size_t)(n0 + r) * DMODEL + k0 + sc]);
            *(unsigned long long*)(&As[r * 40 + sc]) = pack4bf(xa);
            *(unsigned long long*)(&Bs[r * 40 + sc]) = pack4bf(wa);
        }
        if (k0 + 32 < DMODEL) {          // global_prefetch_b8 for next tile
            __builtin_prefetch(&x[(size_t)(m0 + sr) * DMODEL + k0 + 32 + sc], 0, 1);
            __builtin_prefetch(&W[(size_t)(n0 + sr) * DMODEL + k0 + 32 + sc], 0, 1);
        }
        __syncthreads();
        v16bf af[2], bf[4];
#pragma unroll
        for (int i = 0; i < 2; ++i) af[i] = frag_a_lds(As, wm + 16 * i, 0, 40, lane);
#pragma unroll
        for (int j = 0; j < 4; ++j) bf[j] = frag_b_lds(Bs, wn + 16 * j, 0, 40, lane);
#pragma unroll
        for (int i = 0; i < 2; ++i)
#pragma unroll
            for (int j = 0; j < 4; ++j) acc[i][j] = WMMA_BF16(af[i], bf[j], acc[i][j]);
        __syncthreads();
    }

    const int col = lane & 15, hi = lane >> 4;
#pragma unroll
    for (int j = 0; j < 4; ++j) {
        int n = n0 + wn + j * 16 + col;
        float bv = bias[n];
        int part = (n >= 2 * DMODEL) ? 2 : ((n >= DMODEL) ? 1 : 0);  // no idiv
        int dd = n - part * DMODEL;
        int h = dd >> 6, dh = dd & 63;
        unsigned short* dst = (part == 0) ? Qb : ((part == 1) ? Kb : Vb);
        float sc2 = (part == 0) ? (ATT_SCALE * LOG2E) : 1.f;
#pragma unroll
        for (int i = 0; i < 2; ++i)
#pragma unroll
            for (int v = 0; v < 8; ++v) {
                int m = m0 + wm + i * 16 + v + hi * 8;
                int b = m >> 12, s = m & (SEQ - 1);
                float val = (acc[i][j][v] + bv) * sc2;
                dst[(((size_t)b * NHEAD + h) * SEQ + s) * HDIM + dh] = f2bf(val);
            }
    }
}

// ---------------------------------------------------------------------------
// Kernel 2: flash attention. grid = (B*H, S/128), 8 waves; each wave owns 16
// query rows. Per 32-key chunk: K staged via async global->LDS copy, V staged
// transposed via VGPRs; 4 WMMA for S = Q K^T, online softmax with shfl
// reductions, P restaged through wave-private LDS, 4 WMMA for O += P V.
// ---------------------------------------------------------------------------
__global__ __launch_bounds__(256) void attn_kernel(
    const unsigned short* __restrict__ Qb, const unsigned short* __restrict__ Kb,
    const unsigned short* __restrict__ Vb, unsigned short* __restrict__ Ob) {
    __shared__ alignas(16) unsigned short Ks[32 * 72];   // [key][dh]
    __shared__ alignas(16) unsigned short Vt[64 * 40];   // [dh][key]
    __shared__ alignas(16) unsigned short Ps[8][16 * 40];// per-wave [m][key]

    const int tid = threadIdx.x, lane = tid & 31, wave = tid >> 5;
    const int bh = blockIdx.x;
    const int q0 = blockIdx.y * 128 + wave * 16;
    const unsigned short* Qh = Qb + (size_t)bh * SEQ * HDIM;
    const unsigned short* Kh = Kb + (size_t)bh * SEQ * HDIM;
    const unsigned short* Vh = Vb + (size_t)bh * SEQ * HDIM;

    // Q fragments stay in registers for the whole pass (DH=64 -> 2 frags).
    v16bf qa[2];
    {
        int m = q0 + (lane & 15);
        int kb = (lane >> 4) ? 8 : 0;
#pragma unroll
        for (int s = 0; s < 2; ++s) {
            FragU f;
            f.q[0] = *(const uint4*)(Qh + (size_t)m * HDIM + 32 * s + kb);
            f.q[1] = *(const uint4*)(Qh + (size_t)m * HDIM + 32 * s + kb + 16);
            qa[s] = f.v;
        }
    }

    const v8f vzero = {0.f, 0.f, 0.f, 0.f, 0.f, 0.f, 0.f, 0.f};
    v8f acc[4];
#pragma unroll
    for (int j = 0; j < 4; ++j) acc[j] = vzero;
    float mrow[8], lrow[8];
#pragma unroll
    for (int v = 0; v < 8; ++v) { mrow[v] = -1e30f; lrow[v] = 0.f; }

    const int col = lane & 15, hi = lane >> 4;
    const int key = tid >> 3;
    const int dh8 = (tid & 7) * 8;

    for (int c0 = 0; c0 < SEQ; c0 += 32) {
        __syncthreads();  // previous chunk fully consumed by all waves
        {
            // K: async 16B global->LDS copy per lane (32x64 bf16, natural).
            async_copy_b128(&Ks[key * 72 + dh8],
                            Kh + (size_t)(c0 + key) * HDIM + dh8);
            // V: VGPR load + transposed scatter into [dh][key].
            U4H vv;
            vv.q = *(const uint4*)(Vh + (size_t)(c0 + key) * HDIM + dh8);
#pragma unroll
            for (int e = 0; e < 8; ++e) Vt[(dh8 + e) * 40 + key] = vv.h[e];
            wait_asynccnt0();         // K bytes resident in LDS
        }
        __syncthreads();              // (includes dscnt wait for Vt stores)

        // S tiles: cols 0-15 and 16-31 of this key chunk (K over dh = 64).
        v8f s0 = vzero, s1 = vzero;
        s0 = WMMA_BF16(qa[0], frag_b_lds(Ks, 0, 0, 72, lane), s0);
        s0 = WMMA_BF16(qa[1], frag_b_lds(Ks, 0, 32, 72, lane), s0);
        s1 = WMMA_BF16(qa[0], frag_b_lds(Ks, 16, 0, 72, lane), s1);
        s1 = WMMA_BF16(qa[1], frag_b_lds(Ks, 16, 32, 72, lane), s1);

        // Online softmax (base-2; scale*log2e folded into Q).
        float alpha[8];
#pragma unroll
        for (int v = 0; v < 8; ++v) {
            float t = fmaxf(s0[v], s1[v]);
            t = fmaxf(t, __shfl_xor(t, 1, 32));
            t = fmaxf(t, __shfl_xor(t, 2, 32));
            t = fmaxf(t, __shfl_xor(t, 4, 32));
            t = fmaxf(t, __shfl_xor(t, 8, 32));
            float mn = fmaxf(mrow[v], t);
            alpha[v] = exp2f(mrow[v] - mn);
            mrow[v] = mn;
            float p0 = exp2f(s0[v] - mn);
            float p1 = exp2f(s1[v] - mn);
            s0[v] = p0; s1[v] = p1;
            float rs = p0 + p1;
            rs += __shfl_xor(rs, 1, 32);
            rs += __shfl_xor(rs, 2, 32);
            rs += __shfl_xor(rs, 4, 32);
            rs += __shfl_xor(rs, 8, 32);
            lrow[v] = lrow[v] * alpha[v] + rs;
        }
#pragma unroll
        for (int j = 0; j < 4; ++j)
#pragma unroll
            for (int v = 0; v < 8; ++v) acc[j][v] *= alpha[v];

        // Restage P (C layout -> A layout) through wave-private LDS.
        unsigned short* P = &Ps[wave][0];
#pragma unroll
        for (int v = 0; v < 8; ++v) {
            P[(v + hi * 8) * 40 + col]      = f2bf(s0[v]);
            P[(v + hi * 8) * 40 + col + 16] = f2bf(s1[v]);
        }
        // Same-wave LDS ops are in-order; compiler inserts the dscnt wait.
        v16bf pa = frag_a_lds(P, 0, 0, 40, lane);
#pragma unroll
        for (int j = 0; j < 4; ++j)
            acc[j] = WMMA_BF16(pa, frag_b_lds(Vt, j * 16, 0, 40, lane), acc[j]);
    }

    // Epilogue: O = acc / l, stored bf16 at [b, s, h*64 + dh] (== [B,S,D]).
    const int b = bh / NHEAD, h = bh % NHEAD;
#pragma unroll
    for (int j = 0; j < 4; ++j)
#pragma unroll
        for (int v = 0; v < 8; ++v) {
            int s = q0 + v + hi * 8;
            float o = acc[j][v] / lrow[v];
            Ob[((size_t)b * SEQ + s) * DMODEL + h * HDIM + j * 16 + col] = f2bf(o);
        }
}

// ---------------------------------------------------------------------------
// Kernel 3: out = O @ W_out^T + b_out, f32 result. Same tiling as kernel 1;
// A is already bf16 in workspace, so its staging is a pure async LDS copy.
// ---------------------------------------------------------------------------
__global__ __launch_bounds__(256) void out_gemm_kernel(
    const unsigned short* __restrict__ Ob, const float* __restrict__ W,
    const float* __restrict__ bias, float* __restrict__ out) {
    __shared__ alignas(16) unsigned short As[128 * 40];
    __shared__ alignas(16) unsigned short Bs[128 * 40];

    const int tid = threadIdx.x, lane = tid & 31, wave = tid >> 5;
    const int m0 = blockIdx.x * 128, n0 = blockIdx.y * 128;
    const int wm = (wave & 3) * 32, wn = (wave >> 2) * 64;
    const int sr = tid >> 3, sc = (tid & 7) * 4;

    const v8f vzero = {0.f, 0.f, 0.f, 0.f, 0.f, 0.f, 0.f, 0.f};
    v8f acc[2][4];
#pragma unroll
    for (int i = 0; i < 2; ++i)
#pragma unroll
        for (int j = 0; j < 4; ++j) acc[i][j] = vzero;

    for (int k0 = 0; k0 < DMODEL; k0 += 32) {
        {   // A: 128x32 bf16 tile, two async 16B copies per thread
            int r = tid >> 1, c = (tid & 1) * 16;
            const unsigned short* gsrc = Ob + (size_t)(m0 + r) * DMODEL + k0 + c;
            async_copy_b128(&As[r * 40 + c],     gsrc);
            async_copy_b128(&As[r * 40 + c + 8], gsrc + 8);
        }
#pragma unroll
        for (int i = 0; i < 4; ++i) {   // B: W_out f32 -> bf16, float4 + b64
            int r = i * 32 + sr;
            float4 wa = *(const float4*)(&W[(size_t)(n0 + r) * DMODEL + k0 + sc]);
            *(unsigned long long*)(&Bs[r * 40 + sc]) = pack4bf(wa);
        }
        if (k0 + 32 < DMODEL) {
            __builtin_prefetch(&W[(size_t)(n0 + sr) * DMODEL + k0 + 32 + sc], 0, 1);
        }
        wait_asynccnt0();
        __syncthreads();
        v16bf af[2], bf[4];
#pragma unroll
        for (int i = 0; i < 2; ++i) af[i] = frag_a_lds(As, wm + 16 * i, 0, 40, lane);
#pragma unroll
        for (int j = 0; j < 4; ++j) bf[j] = frag_b_lds(Bs, wn + 16 * j, 0, 40, lane);
#pragma unroll
        for (int i = 0; i < 2; ++i)
#pragma unroll
            for (int j = 0; j < 4; ++j) acc[i][j] = WMMA_BF16(af[i], bf[j], acc[i][j]);
        __syncthreads();
    }

    const int col = lane & 15, hi = lane >> 4;
#pragma unroll
    for (int j = 0; j < 4; ++j) {
        int n = n0 + wn + j * 16 + col;
        float bv = bias[n];
#pragma unroll
        for (int i = 0; i < 2; ++i)
#pragma unroll
            for (int v = 0; v < 8; ++v) {
                int m = m0 + wm + i * 16 + v + hi * 8;
                out[(size_t)m * DMODEL + n] = acc[i][j][v] + bv;
            }
    }
}

// ---------------------------------------------------------------------------
extern "C" void kernel_launch(void* const* d_in, const int* in_sizes, int n_in,
                              void* d_out, int out_size, void* d_ws, size_t ws_size,
                              hipStream_t stream) {
    (void)in_sizes; (void)n_in; (void)out_size; (void)ws_size;
    const float* x    = (const float*)d_in[0];
    const float* Wqkv = (const float*)d_in[1];
    const float* bqkv = (const float*)d_in[2];
    const float* Wout = (const float*)d_in[3];
    const float* bout = (const float*)d_in[4];
    float* out = (float*)d_out;

    const size_t plane = (size_t)MROWS * DMODEL;  // 6.29M bf16 elems each
    unsigned short* Qb = (unsigned short*)d_ws;
    unsigned short* Kb = Qb + plane;
    unsigned short* Vb = Kb + plane;
    unsigned short* Ob = Vb + plane;

    qkv_gemm_kernel<<<dim3(MROWS / 128, NQKV / 128), 256, 0, stream>>>(
        x, Wqkv, bqkv, Qb, Kb, Vb);
    attn_kernel<<<dim3(BATCH * NHEAD, SEQ / 128), 256, 0, stream>>>(Qb, Kb, Vb, Ob);
    out_gemm_kernel<<<dim3(MROWS / 128, DMODEL / 128), 256, 0, stream>>>(
        Ob, Wout, bout, out);
}
